// MoE_47425028882671
// MI455X (gfx1250) — matrix operand order repaired and checked by reading, not compiled, not run on hardware
//
#include <hip/hip_runtime.h>
#include <hip/hip_bf16.h>
#include <math.h>

// ---------------------------------------------------------------------------
// MoE fused forward for MI455X (gfx1250, wave32, WMMA 16x16x32 f16 -> f32)
//   B=32768, D=1024, E=16, H1=256, H2=128, GH=128
// Round 2: double-buffered weight staging + async global->LDS copies
// ---------------------------------------------------------------------------

typedef _Float16 h16 __attribute__((ext_vector_type(16)));
typedef float    f8  __attribute__((ext_vector_type(8)));

#define THREADS 256
#define BCONST  32768
#define DCONST  1024
#define ECONST  16
#define H1CONST 256
#define H2CONST 128
#define GHCONST 128

// ---------------------------------------------------------------------------
// Async global -> LDS copy (CDNA5 GLOBAL_LOAD_ASYNC_TO_LDS_B128, ASYNCcnt)
// 16 bytes per lane per issue; completion enforced with s_wait_asynccnt 0
// before the workgroup barrier.
// ---------------------------------------------------------------------------
__device__ __forceinline__ void async_copy16(void* lds, const void* gptr) {
    unsigned l = (unsigned)(size_t)lds;  // generic LDS addr: low 32 bits = LDS offset
    asm volatile("global_load_async_to_lds_b128 %0, %1, off"
                 :: "v"(l), "v"(gptr) : "memory");
}
__device__ __forceinline__ void wait_async0() {
    asm volatile("s_wait_asynccnt 0" ::: "memory");
}

// ---------------------------------------------------------------------------
// Fragment loaders (per CDNA5 ISA §7.12.2 VGPR layouts, wave32)
// A (16x32, f16): lane&15 = M row; slots 0..7 -> K = k0 + (lane>>4)*8 + 0..7
//                 slots 8..15 -> K = k0 + 16 + (lane>>4)*8 + 0..7
// B (32x16, f16) from PRE-TRANSPOSED weights Bt[n][k]:
//                 lane&15 = N col; slots/K mapping same as A
// C/D (16x16, f32): lane&15 = N; VGPR r -> M = (lane>>4)*8 + r
// ---------------------------------------------------------------------------
__device__ __forceinline__ h16 frag_a(const _Float16* t, int m0, int k0, int lda, int lane) {
    const int l15 = lane & 15, hi = lane >> 4;
    const _Float16* p = t + (m0 + l15) * lda + k0 + hi * 8;
    union { uint4 u[2]; h16 v; } r;
    r.u[0] = *(const uint4*)(p);
    r.u[1] = *(const uint4*)(p + 16);
    return r.v;
}

__device__ __forceinline__ h16 frag_b(const _Float16* t, int n0, int k0, int ldb, int lane) {
    const int l15 = lane & 15, hi = lane >> 4;
    const _Float16* p = t + (n0 + l15) * ldb + k0 + hi * 8;
    union { uint4 u[2]; h16 v; } r;
    r.u[0] = *(const uint4*)(p);
    r.u[1] = *(const uint4*)(p + 16);
    return r.v;
}

__device__ __forceinline__ void store_c_relu(const f8& acc, _Float16* t, int m0, int n0,
                                             int ldt, int lane, float bias) {
    const int l15 = lane & 15, hi = lane >> 4;
    const int n = n0 + l15;
#pragma unroll
    for (int r = 0; r < 8; ++r) {
        float v = acc[r] + bias;
        v = v > 0.f ? v : 0.f;
        t[(m0 + hi * 8 + r) * ldt + n] = (_Float16)v;
    }
}

__device__ __forceinline__ f8 wmma_f16(const h16& a, const h16& b, const f8& c) {
    return __builtin_amdgcn_wmma_f32_16x16x32_f16(false, a, false, b, (short)0, c,
                                                  false, false);
}

// ---------------------------------------------------------------------------
// Conversion kernels
// ---------------------------------------------------------------------------
__global__ void cvt_f32_to_f16(const float* __restrict__ in, _Float16* __restrict__ out,
                               long n) {
    long i = (long)blockIdx.x * blockDim.x + threadIdx.x;
    long stride = (long)gridDim.x * blockDim.x;
    for (; i < n; i += stride) out[i] = (_Float16)in[i];
}

// in: [E][R][C] f32 -> out: [E][C][R] f16   (transpose + convert)
__global__ void transpose_cvt(const float* __restrict__ in, _Float16* __restrict__ out,
                              int E, int R, int C) {
    long total = (long)E * R * C;
    long i = (long)blockIdx.x * blockDim.x + threadIdx.x;
    long stride = (long)gridDim.x * blockDim.x;
    for (; i < total; i += stride) {
        long e = i / ((long)R * C);
        long rem = i - e * (long)R * C;
        int r = (int)(rem / C);
        int c = (int)(rem - (long)r * C);
        out[(e * C + c) * R + r] = (_Float16)in[i];
    }
}

// ---------------------------------------------------------------------------
// Gating kernel: g = relu(x @ Wg1 + bg1); logits = g @ Wg2 + bg2; softmax.
// One workgroup = 32 rows. Writes gating weights [B,E] directly to output.
// ---------------------------------------------------------------------------
__global__ __launch_bounds__(256) void gating_kernel(
    const _Float16* __restrict__ xh, const _Float16* __restrict__ Wg1t,
    const float* __restrict__ bg1, const float* __restrict__ Wg2,
    const float* __restrict__ bg2, float* __restrict__ gw) {
    __shared__ _Float16 sX[32 * DCONST];         // 64 KB: x tile
    __shared__ _Float16 sW[2][GHCONST * 32];     // 16 KB: Wg1t K-slices (ping-pong)
    __shared__ _Float16 sG[32 * GHCONST];        //  8 KB: g tile
    __shared__ float    sL[32 * ECONST];         //  2 KB: logits

    const int tid  = threadIdx.x;
    const int lane = tid & 31;
    const int wv   = tid >> 5;
    const int row0 = blockIdx.x * 32;

    // stage x tile (contiguous 64 KB) asynchronously
    {
        const uint4* src = (const uint4*)(xh + (size_t)row0 * DCONST);
        uint4* dst = (uint4*)sX;
        for (int i = tid; i < 32 * DCONST / 8; i += THREADS)
            async_copy16(dst + i, src + i);
    }

    auto stage_wg = [&](int buf, int kk) {
        if (tid < GHCONST) {  // 128 rows x 32 halves (64 B per thread)
            const uint4* src = (const uint4*)(Wg1t + (size_t)tid * DCONST + kk);
            uint4* dst = (uint4*)(sW[buf] + tid * 32);
            async_copy16(dst + 0, src + 0);
            async_copy16(dst + 1, src + 1);
            async_copy16(dst + 2, src + 2);
            async_copy16(dst + 3, src + 3);
        }
    };

    const int mblk = (wv & 1) * 16;
    const int nq   = wv >> 1;  // 0..3 -> n-blocks nq*2, nq*2+1
    f8 acc[2];
    for (int i = 0; i < 2; ++i)
        for (int j = 0; j < 8; ++j) acc[i][j] = 0.f;

    stage_wg(0, 0);
    int pb = 0;
    for (int kk = 0; kk < DCONST; kk += 32, pb ^= 1) {
        wait_async0();
        __syncthreads();  // slice pb ready; everyone done with slice pb^1
        if (kk + 32 < DCONST) stage_wg(pb ^ 1, kk + 32);
        h16 a = frag_a(sX, mblk, kk, DCONST, lane);
#pragma unroll
        for (int i = 0; i < 2; ++i) {
            h16 b = frag_b(sW[pb], (nq * 2 + i) * 16, 0, 32, lane);
            acc[i] = wmma_f16(a, b, acc[i]);
        }
    }
    {
        const int l15 = lane & 15;
#pragma unroll
        for (int i = 0; i < 2; ++i) {
            const int n0 = (nq * 2 + i) * 16;
            store_c_relu(acc[i], sG, mblk, n0, GHCONST, lane, bg1[n0 + l15]);
        }
    }
    __syncthreads();

    // logits: [32][16], 512 dots of K=128
    for (int idx = tid; idx < 32 * ECONST; idx += THREADS) {
        const int m = idx >> 4, e = idx & 15;
        float s = bg2[e];
        for (int k = 0; k < GHCONST; ++k)
            s += (float)sG[m * GHCONST + k] * Wg2[k * ECONST + e];
        sL[idx] = s;
    }
    __syncthreads();

    // softmax per row
    if (tid < 32) {
        const int m = tid;
        float mx = sL[m * 16];
        for (int e = 1; e < 16; ++e) mx = fmaxf(mx, sL[m * 16 + e]);
        float ex[16], sum = 0.f;
        for (int e = 0; e < 16; ++e) { ex[e] = __expf(sL[m * 16 + e] - mx); sum += ex[e]; }
        const float inv = 1.f / sum;
        for (int e = 0; e < 16; ++e)
            gw[(size_t)(row0 + m) * ECONST + e] = ex[e] * inv;
    }
}

// ---------------------------------------------------------------------------
// Fused expert kernel: one workgroup = 32 rows x 1 expert; all 3 layers.
// Intermediates (h1, h2) never leave LDS.
// ---------------------------------------------------------------------------
__global__ __launch_bounds__(256) void moe_expert_kernel(
    const _Float16* __restrict__ xh, const _Float16* __restrict__ W1t,
    const float* __restrict__ b1, const _Float16* __restrict__ W2t,
    const float* __restrict__ b2, const float* __restrict__ W3,
    const float* __restrict__ b3, float* __restrict__ eo) {
    __shared__ _Float16 s0[32 * DCONST];         // 64 KB: x tile, then W2t[e] full
    __shared__ _Float16 sW[2][H1CONST * 32];     // 32 KB: W1t K-slices (ping-pong)
    __shared__ _Float16 sH1[32 * H1CONST];       // 16 KB
    __shared__ _Float16 sH2[32 * H2CONST];       //  8 KB

    const int tid  = threadIdx.x;
    const int lane = tid & 31;
    const int wv   = tid >> 5;
    const int e    = blockIdx.y;
    const int row0 = blockIdx.x * 32;

    // stage x tile (contiguous 64 KB) asynchronously
    {
        const uint4* src = (const uint4*)(xh + (size_t)row0 * DCONST);
        uint4* dst = (uint4*)s0;
        for (int i = tid; i < 32 * DCONST / 8; i += THREADS)
            async_copy16(dst + i, src + i);
    }

    auto stage_w1 = [&](int buf, int kk) {  // 256 rows x 32 halves (64 B per thread)
        const uint4* src =
            (const uint4*)(W1t + ((size_t)(e * H1CONST + tid)) * DCONST + kk);
        uint4* dst = (uint4*)(sW[buf] + tid * 32);
        async_copy16(dst + 0, src + 0);
        async_copy16(dst + 1, src + 1);
        async_copy16(dst + 2, src + 2);
        async_copy16(dst + 3, src + 3);
    };

    // ---- layer 1: h1[32][256] = relu(x @ W1[e] + b1[e]) ----
    const int mblk = (wv & 1) * 16;
    const int nq   = wv >> 1;  // 0..3 -> n-blocks nq*4 .. nq*4+3
    f8 acc[4];
    for (int i = 0; i < 4; ++i)
        for (int j = 0; j < 8; ++j) acc[i][j] = 0.f;

    stage_w1(0, 0);
    int pb = 0;
    for (int kk = 0; kk < DCONST; kk += 32, pb ^= 1) {
        wait_async0();
        __syncthreads();  // slice pb ready; everyone done with slice pb^1
        if (kk + 32 < DCONST) stage_w1(pb ^ 1, kk + 32);
        h16 a = frag_a(s0, mblk, kk, DCONST, lane);
#pragma unroll
        for (int i = 0; i < 4; ++i) {
            h16 b = frag_b(sW[pb], (nq * 4 + i) * 16, 0, 32, lane);
            acc[i] = wmma_f16(a, b, acc[i]);
        }
    }
    {
        const int l15 = lane & 15;
#pragma unroll
        for (int i = 0; i < 4; ++i) {
            const int n0 = (nq * 4 + i) * 16;
            store_c_relu(acc[i], sH1, mblk, n0, H1CONST, lane,
                         b1[e * H1CONST + n0 + l15]);
        }
    }
    __syncthreads();  // layer-1 reads of s0 (x tile) complete

    // ---- layer 2: h2[32][128] = relu(h1 @ W2[e] + b2[e]) ----
    // stage full W2t[e] (128x256 halves == 64 KB) into s0 (x tile dead)
    {
        const uint4* src = (const uint4*)(W2t + (size_t)e * H2CONST * H1CONST);
        uint4* dst = (uint4*)s0;
        for (int i = tid; i < H2CONST * H1CONST / 8; i += THREADS)
            async_copy16(dst + i, src + i);
    }
    wait_async0();
    __syncthreads();

    const int nq2 = wv >> 1;  // 0..3 -> n-blocks nq2*2, nq2*2+1
    f8 acc2[2];
    for (int i = 0; i < 2; ++i)
        for (int j = 0; j < 8; ++j) acc2[i][j] = 0.f;

    for (int k = 0; k < H1CONST; k += 32) {
        h16 a = frag_a(sH1, mblk, k, H1CONST, lane);
#pragma unroll
        for (int i = 0; i < 2; ++i) {
            h16 b = frag_b(s0, (nq2 * 2 + i) * 16, k, H1CONST, lane);
            acc2[i] = wmma_f16(a, b, acc2[i]);
        }
    }
    {
        const int l15 = lane & 15;
#pragma unroll
        for (int i = 0; i < 2; ++i) {
            const int n0 = (nq2 * 2 + i) * 16;
            store_c_relu(acc2[i], sH2, mblk, n0, H2CONST, lane,
                         b2[e * H2CONST + n0 + l15]);
        }
    }
    __syncthreads();

    // ---- layer 3: eo[b][e] = h2 . W3[e] + b3[e] ----
    if (tid < 32) {
        float s = b3[e];
        const float* w3 = W3 + e * H2CONST;
        for (int k = 0; k < H2CONST; ++k)
            s += (float)sH2[tid * H2CONST + k] * w3[k];
        eo[(size_t)(row0 + tid) * ECONST + e] = s;
    }
}

// ---------------------------------------------------------------------------
// Combine: final[b] = sum_e gw[b][e] * eo[b][e]
// ---------------------------------------------------------------------------
__global__ void combine_kernel(const float* __restrict__ gw, const float* __restrict__ eo,
                               float* __restrict__ outp) {
    const int b = blockIdx.x * blockDim.x + threadIdx.x;
    if (b < BCONST) {
        float s = 0.f;
#pragma unroll
        for (int e = 0; e < ECONST; ++e)
            s += gw[(size_t)b * ECONST + e] * eo[(size_t)b * ECONST + e];
        outp[b] = s;
    }
}

// ---------------------------------------------------------------------------
// Workspace layout (bytes)
// ---------------------------------------------------------------------------
static const size_t XH_OFF   = 0;            // B*D f16      = 67,108,864
static const size_t W1T_OFF  = 67108864;     // E*H1*D f16   =  8,388,608
static const size_t W2T_OFF  = 75497472;     // E*H2*H1 f16  =  1,048,576
static const size_t WG1T_OFF = 76546048;     // GH*D f16     =    262,144
static const size_t EO_OFF   = 76808192;     // B*E f32      =  2,097,152

extern "C" void kernel_launch(void* const* d_in, const int* in_sizes, int n_in,
                              void* d_out, int out_size, void* d_ws, size_t ws_size,
                              hipStream_t stream) {
    const float* x   = (const float*)d_in[0];
    const float* W1  = (const float*)d_in[1];
    const float* b1  = (const float*)d_in[2];
    const float* W2  = (const float*)d_in[3];
    const float* b2  = (const float*)d_in[4];
    const float* W3  = (const float*)d_in[5];
    const float* b3  = (const float*)d_in[6];
    const float* Wg1 = (const float*)d_in[7];
    const float* bg1 = (const float*)d_in[8];
    const float* Wg2 = (const float*)d_in[9];
    const float* bg2 = (const float*)d_in[10];
    float* outp = (float*)d_out;  // [0,B): final, [B, B+B*E): gating weights

    char* ws = (char*)d_ws;
    _Float16* xh   = (_Float16*)(ws + XH_OFF);
    _Float16* W1t  = (_Float16*)(ws + W1T_OFF);
    _Float16* W2t  = (_Float16*)(ws + W2T_OFF);
    _Float16* Wg1t = (_Float16*)(ws + WG1T_OFF);
    float*    eo   = (float*)(ws + EO_OFF);

    // 1) precision conversion + weight transposes (f32 -> f16)
    cvt_f32_to_f16<<<2048, 256, 0, stream>>>(x, xh, (long)BCONST * DCONST);
    transpose_cvt<<<2048, 256, 0, stream>>>(W1, W1t, ECONST, DCONST, H1CONST);
    transpose_cvt<<<512, 256, 0, stream>>>(W2, W2t, ECONST, H1CONST, H2CONST);
    transpose_cvt<<<128, 256, 0, stream>>>(Wg1, Wg1t, 1, DCONST, GHCONST);

    // 2) gating network (WMMA) -> gating weights straight into d_out
    gating_kernel<<<BCONST / 32, THREADS, 0, stream>>>(xh, Wg1t, bg1, Wg2, bg2,
                                                       outp + BCONST);

    // 3) fused 3-layer experts (WMMA) -> per-expert scalars
    moe_expert_kernel<<<dim3(BCONST / 32, ECONST), THREADS, 0, stream>>>(
        xh, W1t, b1, W2t, b2, W3, b3, eo);

    // 4) weighted combine
    combine_kernel<<<BCONST / 256, 256, 0, stream>>>(outp + BCONST, eo, outp);
}
